// message_passing_gnn_induct_77403900609206
// MI455X (gfx1250) — compile-verified, matrix operand order checked by reading, not run on hardware
//
#include <hip/hip_runtime.h>

#define NN 50000
#define EE 800000
#define DD 64

typedef __bf16 bf16;
typedef __attribute__((ext_vector_type(16))) __bf16 v16bf;
typedef __attribute__((ext_vector_type(8)))  float  v8f;
typedef __attribute__((ext_vector_type(4)))  unsigned int u32x4;

// GCC-style int4 matching the async-LDS builtin's canonical parameter type.
typedef int i32x4v __attribute__((vector_size(16)));
typedef __attribute__((address_space(1))) i32x4v as1_i32x4;
typedef __attribute__((address_space(3))) i32x4v as3_i32x4;

union TileBF { v16bf v; u32x4 q[2]; };

#define HAS_ASYNC_LDS (__has_builtin(__builtin_amdgcn_global_load_async_to_lds_b128) && \
                       __has_builtin(__builtin_amdgcn_s_wait_asynccnt))

// ---------------------------------------------------------------------------
// WMMA helper: D(16x16 f32) += A(16x32 bf16) x B(32x16 bf16)
// ---------------------------------------------------------------------------
__device__ __forceinline__ v8f wmma_bf16(v16bf a, v16bf b, v8f c) {
  return __builtin_amdgcn_wmma_f32_16x16x32_bf16(false, a, false, b, (short)0, c,
                                                 false, false);
}

// A-operand layout (16-bit A 16x32): lane holds row (lane&15); elements 0..7 ->
// K = h8+0..7, elements 8..15 -> K = 16+h8+0..7, where h8 = (lane>=16)?8:0.
__device__ __forceinline__ v16bf load_a_row(const bf16* p, int lane) {
  TileBF t; const int h8 = (lane >> 4) << 3;
  t.q[0] = *(const u32x4*)(p + h8);
  t.q[1] = *(const u32x4*)(p + 16 + h8);
  return t.v;
}

// B-operand from permuted weights: slot = ntile*KS + kslab; (slot,lane) owns
// 16 contiguous bf16 (32B) -> two b128 loads (global or LDS depending on ptr).
__device__ __forceinline__ v16bf load_b_perm(const bf16* Wp, int slot, int lane) {
  const bf16* p = Wp + (((size_t)(slot << 5) + lane) << 4);
  TileBF t;
  t.q[0] = *(const u32x4*)(p);
  t.q[1] = *(const u32x4*)(p + 8);
  return t.v;
}

// bf16 A-slab from bf16 row with per-channel affine (BN) + ReLU.
__device__ __forceinline__ v16bf load_a_bf16_affine_relu(const bf16* p, int lane,
                                                         const float* sc, const float* sh,
                                                         int cbase) {
  TileBF t; const int h8 = (lane >> 4) << 3;
  t.q[0] = *(const u32x4*)(p + h8);
  t.q[1] = *(const u32x4*)(p + 16 + h8);
  v16bf a;
#pragma unroll
  for (int e = 0; e < 16; ++e) {
    const int kl = h8 + (e & 7) + ((e >> 3) << 4);
    const int c = cbase + kl;
    const float v = fmaxf((float)t.v[e] * sc[c] + sh[c], 0.f);
    a[e] = (bf16)v;
  }
  return a;
}

// bf16 A-slab from f32 row scaled by scalar (degree-norm).
__device__ __forceinline__ v16bf load_a_scaled(const float* p, int lane, float s) {
  v16bf a; const int h8 = (lane >> 4) << 3;
#pragma unroll
  for (int e = 0; e < 16; ++e) {
    const int kl = h8 + (e & 7) + ((e >> 3) << 4);
    a[e] = (bf16)(p[kl] * s);
  }
  return a;
}

// Stage a permuted-weight block into LDS: async global->LDS copy when the
// toolchain exposes it (global_load_async_to_lds_b128 + s_wait_asynccnt),
// otherwise a plain b128 copy. Caller must __syncthreads() afterwards.
__device__ __forceinline__ void stage_w(const bf16* __restrict__ gw, bf16* sw,
                                        int nelem, int tid) {
#if HAS_ASYNC_LDS
  for (int i = tid * 8; i < nelem; i += 256 * 8) {
    __builtin_amdgcn_global_load_async_to_lds_b128(
        (as1_i32x4*)(gw + i), (as3_i32x4*)(sw + i), 0, 0);
  }
  __builtin_amdgcn_s_wait_asynccnt(0);
#else
  for (int i = tid * 8; i < nelem; i += 256 * 8)
    *(u32x4*)(sw + i) = *(const u32x4*)(gw + i);
#endif
}

// ---------------------------------------------------------------------------
// Weight permutation: W [K,N] f32 row-major -> bf16 buffer in B-lane order.
// ---------------------------------------------------------------------------
__global__ void k_permw(const float* __restrict__ W, bf16* __restrict__ Wp, int K, int N) {
  const int o = blockIdx.x * blockDim.x + threadIdx.x;
  if (o < K * N) {
    const int e = o & 15, l = (o >> 4) & 31, slot = o >> 9;
    const int KS = K >> 5;
    const int s = slot % KS, t = slot / KS;
    const int k = (s << 5) + ((l >> 4) << 3) + (e & 7) + ((e >> 3) << 4);
    const int n = (t << 4) + (l & 15);
    Wp[o] = (bf16)W[(size_t)k * N + n];
  }
}

__global__ void k_embed(const int* __restrict__ nodes, const float* __restrict__ emb,
                        float* __restrict__ x, bf16* __restrict__ xb) {
  const int i = blockIdx.x * blockDim.x + threadIdx.x;
  if (i < NN * DD) {
    const int n = i >> 6, c = i & 63;
    const float v = emb[(size_t)nodes[n] * DD + c];
    x[i] = v; xb[i] = (bf16)v;
  }
}

__global__ void k_deg(const int* __restrict__ dst, const int* __restrict__ src,
                      float* __restrict__ dP, float* __restrict__ dC) {
  const int i = blockIdx.x * blockDim.x + threadIdx.x;
  if (i < EE) { atomicAdd(dP + dst[i], 1.f); atomicAdd(dC + src[i], 1.f); }
}

__global__ void k_dinv(float* __restrict__ dP, float* __restrict__ dC) {
  const int i = blockIdx.x * blockDim.x + threadIdx.x;
  if (i < NN) {
    const float a = dP[i]; dP[i] = (a > 0.f) ? 1.f / a : 0.f;
    const float b = dC[i]; dC[i] = (b > 0.f) ? 1.f / b : 0.f;
  }
}

// ---------------------------------------------------------------------------
// Edge GEMM1: h[e,:] = concat(xb[i], xb[j]) @ W1 + b1 (K=128), h stored bf16.
// Fused BN-stat reduction in f32 from the exact f32 accumulators.
// ---------------------------------------------------------------------------
__global__ void k_edge_gemm1(const bf16* __restrict__ xb,
                             const int* __restrict__ idx_i, const int* __restrict__ idx_j,
                             const bf16* __restrict__ Wp, const float* __restrict__ bias,
                             bf16* __restrict__ hout,
                             float* __restrict__ gsum, float* __restrict__ gsq) {
  __shared__ bf16  sW[128 * 64];    // 16 KB staged weights
  __shared__ float lsum[DD];
  __shared__ float lsq[DD];
  const int tid = threadIdx.x;
  if (tid < DD) { lsum[tid] = 0.f; lsq[tid] = 0.f; }
  stage_w(Wp, sW, 128 * 64, tid);
  __syncthreads();
  const int lane = tid & 31;
  const int wv = blockIdx.x * (blockDim.x >> 5) + (tid >> 5);
  const int e0 = wv << 4;
  if (e0 < EE) {
    const int r = lane & 15;
    const int ro = (lane >> 4) << 3;
    const int ii = idx_i[e0 + r];
    const int jj = idx_j[e0 + r];
    const bf16* pi = xb + (size_t)ii * DD;
    const bf16* pj = xb + (size_t)jj * DD;
    v16bf a[4];
    a[0] = load_a_row(pi, lane);      a[1] = load_a_row(pi + 32, lane);
    a[2] = load_a_row(pj, lane);      a[3] = load_a_row(pj + 32, lane);
#pragma unroll
    for (int t = 0; t < 4; ++t) {
      v8f acc;
#pragma unroll
      for (int g = 0; g < 8; ++g) acc[g] = 0.f;
#pragma unroll
      for (int s = 0; s < 4; ++s)
        acc = wmma_bf16(a[s], load_b_perm(sW, t * 4 + s, lane), acc);
      const int c = (t << 4) + r;
      const float bv = bias[c];
      float s1 = 0.f, s2 = 0.f;
#pragma unroll
      for (int g = 0; g < 8; ++g) {
        const float v = acc[g] + bv;
        hout[(size_t)(e0 + ro + g) * DD + c] = (bf16)v;
        s1 += v; s2 += v * v;
      }
      s1 += __shfl_xor(s1, 16, 32);
      s2 += __shfl_xor(s2, 16, 32);
      if (lane < 16) { atomicAdd(&lsum[c], s1); atomicAdd(&lsq[c], s2); }
    }
  }
  __syncthreads();
  if (tid < DD) { atomicAdd(gsum + tid, lsum[tid]); atomicAdd(gsq + tid, lsq[tid]); }
}

// BN finalize: scale/shift from accumulated sums (biased variance).
__global__ void k_bnfin(const float* __restrict__ gsum, const float* __restrict__ gsq,
                        const float* __restrict__ gamma, const float* __restrict__ beta,
                        float invE, float* __restrict__ sc, float* __restrict__ sh) {
  const int c = threadIdx.x;
  if (c < DD) {
    const float mu = gsum[c] * invE;
    const float var = gsq[c] * invE - mu * mu;
    const float rs = rsqrtf(var + 1e-5f);
    const float s = gamma[c] * rs;
    sc[c] = s;
    sh[c] = beta[c] - mu * s;
  }
}

// ---------------------------------------------------------------------------
// Edge GEMM2 (in place, bf16): h = relu(BN1(h)) @ W2 + b2, fused BN2 stats.
// ---------------------------------------------------------------------------
__global__ void k_edge_gemm2(bf16* __restrict__ h, const bf16* __restrict__ Wp,
                             const float* __restrict__ sc, const float* __restrict__ sh,
                             const float* __restrict__ bias,
                             float* __restrict__ gsum, float* __restrict__ gsq) {
  __shared__ bf16  sW[64 * 64];     // 8 KB staged weights
  __shared__ float lsum[DD];
  __shared__ float lsq[DD];
  __shared__ float ssc[DD];
  __shared__ float ssh[DD];
  const int tid = threadIdx.x;
  if (tid < DD) { lsum[tid] = 0.f; lsq[tid] = 0.f; ssc[tid] = sc[tid]; ssh[tid] = sh[tid]; }
  stage_w(Wp, sW, 64 * 64, tid);
  __syncthreads();
  const int lane = tid & 31;
  const int wv = blockIdx.x * (blockDim.x >> 5) + (tid >> 5);
  const int e0 = wv << 4;
  if (e0 < EE) {
    const int r = lane & 15;
    const int ro = (lane >> 4) << 3;
    const bf16* prow = h + (size_t)(e0 + r) * DD;
    v16bf a[2];
    a[0] = load_a_bf16_affine_relu(prow, lane, ssc, ssh, 0);
    a[1] = load_a_bf16_affine_relu(prow + 32, lane, ssc, ssh, 32);
#pragma unroll
    for (int t = 0; t < 4; ++t) {
      v8f acc;
#pragma unroll
      for (int g = 0; g < 8; ++g) acc[g] = 0.f;
#pragma unroll
      for (int s = 0; s < 2; ++s)
        acc = wmma_bf16(a[s], load_b_perm(sW, t * 2 + s, lane), acc);
      const int c = (t << 4) + r;
      const float bv = bias[c];
      float s1 = 0.f, s2 = 0.f;
#pragma unroll
      for (int g = 0; g < 8; ++g) {
        const float v = acc[g] + bv;
        h[(size_t)(e0 + ro + g) * DD + c] = (bf16)v;
        s1 += v; s2 += v * v;
      }
      s1 += __shfl_xor(s1, 16, 32);
      s2 += __shfl_xor(s2, 16, 32);
      if (lane < 16) { atomicAdd(&lsum[c], s1); atomicAdd(&lsq[c], s2); }
    }
  }
  __syncthreads();
  if (tid < DD) { atomicAdd(gsum + tid, lsum[tid]); atomicAdd(gsq + tid, lsq[tid]); }
}

// Scatter: agg[idx[e],c] += relu(BN2(h[e,c]))  (f32 atomics)
__global__ void k_scatter(const bf16* __restrict__ h, const float* __restrict__ sc,
                          const float* __restrict__ sh, const int* __restrict__ idx,
                          float* __restrict__ agg) {
  const int i = blockIdx.x * blockDim.x + threadIdx.x;
  if (i < EE * DD) {
    const int e = i >> 6, c = i & 63;
    const float v = fmaxf((float)h[i] * sc[c] + sh[c], 0.f);
    atomicAdd(agg + (size_t)idx[e] * DD + c, v);
  }
}

// ---------------------------------------------------------------------------
// Node update: upd = relu([x, fi, fo] @ fcW + fcb) @ fc2W + fc2b ; x += upd
// Stage1 output transposed through LDS back into WMMA A-layout for stage2.
// ---------------------------------------------------------------------------
__global__ void k_node_update(float* __restrict__ x, bf16* __restrict__ xb,
                              const float* __restrict__ aggP, const float* __restrict__ dinvP,
                              const float* __restrict__ aggC, const float* __restrict__ dinvC,
                              const bf16* __restrict__ fcWp, const float* __restrict__ fcb,
                              const bf16* __restrict__ fc2Wp, const float* __restrict__ fc2b) {
  __shared__ bf16 lt[8][16 * 136];   // per-wave 16x128 tile, pitch 136 (pad)
  const int tid = threadIdx.x, lane = tid & 31, wl = tid >> 5;
  const int wv = blockIdx.x * 8 + wl;
  const int n0 = wv << 4;
  if (n0 >= NN) return;              // wave-uniform
  const int r = lane & 15, ro = (lane >> 4) << 3;
  const int nr = n0 + r;
  const bf16* px = xb + (size_t)nr * DD;
  const float* pp = aggP + (size_t)nr * DD;
  const float* pc = aggC + (size_t)nr * DD;
  const float dp = dinvP[nr], dc = dinvC[nr];
  v16bf a[6];
  a[0] = load_a_row(px, lane);         a[1] = load_a_row(px + 32, lane);
  a[2] = load_a_scaled(pp, lane, dp);  a[3] = load_a_scaled(pp + 32, lane, dp);
  a[4] = load_a_scaled(pc, lane, dc);  a[5] = load_a_scaled(pc + 32, lane, dc);
  bf16* T = &lt[wl][0];
#pragma unroll
  for (int t = 0; t < 8; ++t) {
    v8f acc;
#pragma unroll
    for (int g = 0; g < 8; ++g) acc[g] = 0.f;
#pragma unroll
    for (int s = 0; s < 6; ++s)
      acc = wmma_bf16(a[s], load_b_perm(fcWp, t * 6 + s, lane), acc);
    const int c = (t << 4) + r;
    const float bv = fcb[c];
#pragma unroll
    for (int g = 0; g < 8; ++g)
      T[(ro + g) * 136 + c] = (bf16)fmaxf(acc[g] + bv, 0.f);
  }
  // reload in A layout (per-wave tile -> ds_load_b128; in-wave DS ordering)
  v16bf a2[4];
  const int h8 = (lane >> 4) << 3;
#pragma unroll
  for (int s = 0; s < 4; ++s) {
    const bf16* p = T + r * 136 + s * 32;
    TileBF tt;
    tt.q[0] = *(const u32x4*)(p + h8);
    tt.q[1] = *(const u32x4*)(p + 16 + h8);
    a2[s] = tt.v;
  }
#pragma unroll
  for (int t = 0; t < 4; ++t) {
    v8f acc;
#pragma unroll
    for (int g = 0; g < 8; ++g) acc[g] = 0.f;
#pragma unroll
    for (int s = 0; s < 4; ++s)
      acc = wmma_bf16(a2[s], load_b_perm(fc2Wp, t * 4 + s, lane), acc);
    const int c = (t << 4) + r;
    const float bv = fc2b[c];
#pragma unroll
    for (int g = 0; g < 8; ++g) {
      const size_t idx = (size_t)(n0 + ro + g) * DD + c;
      const float nv = x[idx] + acc[g] + bv;
      x[idx] = nv; xb[idx] = (bf16)nv;
    }
  }
}

// Final: out = x @ conv_W + conv_b  ([N,64] -> [N,128])
__global__ void k_final(const bf16* __restrict__ xb, const bf16* __restrict__ Wp,
                        const float* __restrict__ bias, float* __restrict__ out) {
  const int tid = threadIdx.x, lane = tid & 31;
  const int wv = blockIdx.x * (blockDim.x >> 5) + (tid >> 5);
  const int n0 = wv << 4;
  if (n0 >= NN) return;
  const int r = lane & 15, ro = (lane >> 4) << 3;
  const bf16* prow = xb + (size_t)(n0 + r) * DD;
  const v16bf a0 = load_a_row(prow, lane);
  const v16bf a1 = load_a_row(prow + 32, lane);
#pragma unroll
  for (int t = 0; t < 8; ++t) {
    v8f acc;
#pragma unroll
    for (int g = 0; g < 8; ++g) acc[g] = 0.f;
    acc = wmma_bf16(a0, load_b_perm(Wp, t * 2 + 0, lane), acc);
    acc = wmma_bf16(a1, load_b_perm(Wp, t * 2 + 1, lane), acc);
    const int c = (t << 4) + r;
    const float bv = bias[c];
#pragma unroll
    for (int g = 0; g < 8; ++g)
      out[(size_t)(n0 + ro + g) * 128 + c] = acc[g] + bv;
  }
}

// ---------------------------------------------------------------------------
extern "C" void kernel_launch(void* const* d_in, const int* in_sizes, int n_in,
                              void* d_out, int out_size, void* d_ws, size_t ws_size,
                              hipStream_t stream) {
  (void)in_sizes; (void)n_in; (void)out_size; (void)ws_size;
  const int*   nodes = (const int*)d_in[0];
  const int*   edges = (const int*)d_in[1];
  const int*   esrc  = edges;          // edges[0]
  const int*   edst  = edges + EE;     // edges[1]
  const float* emb   = (const float*)d_in[2];
  const float* p_W1  = (const float*)d_in[3];
  const float* p_b1  = (const float*)d_in[4];
  const float* p_g1  = (const float*)d_in[5];
  const float* p_be1 = (const float*)d_in[6];
  const float* p_W2  = (const float*)d_in[7];
  const float* p_b2  = (const float*)d_in[8];
  const float* p_g2  = (const float*)d_in[9];
  const float* p_be2 = (const float*)d_in[10];
  const float* c_W1  = (const float*)d_in[11];
  const float* c_b1  = (const float*)d_in[12];
  const float* c_g1  = (const float*)d_in[13];
  const float* c_be1 = (const float*)d_in[14];
  const float* c_W2  = (const float*)d_in[15];
  const float* c_b2  = (const float*)d_in[16];
  const float* c_g2  = (const float*)d_in[17];
  const float* c_be2 = (const float*)d_in[18];
  const float* fc_W  = (const float*)d_in[19];
  const float* fc_b  = (const float*)d_in[20];
  const float* fc2_W = (const float*)d_in[21];
  const float* fc2_b = (const float*)d_in[22];
  const float* cv_W  = (const float*)d_in[23];
  const float* cv_b  = (const float*)d_in[24];

  char* w = (char*)d_ws;
  size_t off = 0;
  auto take = [&](size_t bytes) -> void* {
    void* p = w + off;
    off = (off + bytes + 255) & ~(size_t)255;
    return p;
  };
  float* x     = (float*)take((size_t)NN * DD * 4);
  bf16*  xb    = (bf16*) take((size_t)NN * DD * 2);
  bf16*  h     = (bf16*) take((size_t)EE * DD * 2);   // bf16 hidden: halves HBM stream
  float* aggP  = (float*)take((size_t)NN * DD * 4);
  float* aggC  = (float*)take((size_t)NN * DD * 4);
  float* dinvP = (float*)take((size_t)NN * 4);
  float* dinvC = (float*)take((size_t)NN * 4);
  float* fsum  = (float*)take(2 * DD * 4);  float* fsq = fsum + DD;
  float* scA   = (float*)take(2 * DD * 4);  float* shA = scA + DD;
  float* scB   = (float*)take(2 * DD * 4);  float* shB = scB + DD;
  bf16* pW1p  = (bf16*)take(128 * 64 * 2);
  bf16* pW2p  = (bf16*)take(64 * 64 * 2);
  bf16* cW1p  = (bf16*)take(128 * 64 * 2);
  bf16* cW2p  = (bf16*)take(64 * 64 * 2);
  bf16* fcWp  = (bf16*)take(192 * 128 * 2);
  bf16* fc2Wp = (bf16*)take(128 * 64 * 2);
  bf16* cvWp  = (bf16*)take(64 * 128 * 2);

  // weight permutation to bf16 B-lane layout
  k_permw<<<(128 * 64 + 255) / 256, 256, 0, stream>>>(p_W1, pW1p, 128, 64);
  k_permw<<<(64 * 64 + 255) / 256, 256, 0, stream>>>(p_W2, pW2p, 64, 64);
  k_permw<<<(128 * 64 + 255) / 256, 256, 0, stream>>>(c_W1, cW1p, 128, 64);
  k_permw<<<(64 * 64 + 255) / 256, 256, 0, stream>>>(c_W2, cW2p, 64, 64);
  k_permw<<<(192 * 128 + 255) / 256, 256, 0, stream>>>(fc_W, fcWp, 192, 128);
  k_permw<<<(128 * 64 + 255) / 256, 256, 0, stream>>>(fc2_W, fc2Wp, 128, 64);
  k_permw<<<(64 * 128 + 255) / 256, 256, 0, stream>>>(cv_W, cvWp, 64, 128);

  k_embed<<<(NN * DD + 255) / 256, 256, 0, stream>>>(nodes, emb, x, xb);

  (void)hipMemsetAsync(dinvP, 0, (size_t)NN * 4, stream);
  (void)hipMemsetAsync(dinvC, 0, (size_t)NN * 4, stream);
  k_deg<<<(EE + 255) / 256, 256, 0, stream>>>(edst, esrc, dinvP, dinvC);
  k_dinv<<<(NN + 255) / 256, 256, 0, stream>>>(dinvP, dinvC);

  const float invE = 1.0f / (float)EE;
  const int egrid = EE / 128;             // 6250 blocks, 8 waves x 16 edges
  const int ngrid = (NN / 16 + 7) / 8;    // 391 blocks of node tiles

  for (int it = 0; it < 2; ++it) {
    // Parent aggregation: idx_i = dst, idx_j = src, scatter to dst
    (void)hipMemsetAsync(fsum, 0, 2 * DD * 4, stream);
    k_edge_gemm1<<<egrid, 256, 0, stream>>>(xb, edst, esrc, pW1p, p_b1, h, fsum, fsq);
    k_bnfin<<<1, DD, 0, stream>>>(fsum, fsq, p_g1, p_be1, invE, scA, shA);
    (void)hipMemsetAsync(fsum, 0, 2 * DD * 4, stream);
    k_edge_gemm2<<<egrid, 256, 0, stream>>>(h, pW2p, scA, shA, p_b2, fsum, fsq);
    k_bnfin<<<1, DD, 0, stream>>>(fsum, fsq, p_g2, p_be2, invE, scB, shB);
    (void)hipMemsetAsync(aggP, 0, (size_t)NN * DD * 4, stream);
    k_scatter<<<(EE * DD) / 256, 256, 0, stream>>>(h, scB, shB, edst, aggP);

    // Child aggregation: idx_i = src, idx_j = dst, scatter to src
    (void)hipMemsetAsync(fsum, 0, 2 * DD * 4, stream);
    k_edge_gemm1<<<egrid, 256, 0, stream>>>(xb, esrc, edst, cW1p, c_b1, h, fsum, fsq);
    k_bnfin<<<1, DD, 0, stream>>>(fsum, fsq, c_g1, c_be1, invE, scA, shA);
    (void)hipMemsetAsync(fsum, 0, 2 * DD * 4, stream);
    k_edge_gemm2<<<egrid, 256, 0, stream>>>(h, cW2p, scA, shA, c_b2, fsum, fsq);
    k_bnfin<<<1, DD, 0, stream>>>(fsum, fsq, c_g2, c_be2, invE, scB, shB);
    (void)hipMemsetAsync(aggC, 0, (size_t)NN * DD * 4, stream);
    k_scatter<<<(EE * DD) / 256, 256, 0, stream>>>(h, scB, shB, esrc, aggC);

    k_node_update<<<ngrid, 256, 0, stream>>>(x, xb, aggP, dinvP, aggC, dinvC,
                                             fcWp, fc_b, fc2Wp, fc2_b);
  }

  k_final<<<ngrid, 256, 0, stream>>>(xb, cvWp, cv_b, (float*)d_out);
}
